// SSIMLoss_60275571032412
// MI455X (gfx1250) — matrix operand order repaired and checked by reading, not compiled, not run on hardware
//
#include <hip/hip_runtime.h>

// ---------------------------------------------------------------------------
// SSIM loss, fused, MI455X (gfx1250).
//   pred:   [32, 8, 512, 512] f32      target: [32, 512, 512] i64
//   out:    scalar f32 = 1 - mean_bc( relu( mean_spatial( ssim_map ) ) )
// Separable 11-tap Gaussian conv done as banded matmuls on the WMMA pipe
// (V_WMMA_F32_16X16X32_F16, f16 operands / f32 accumulation), one WMMA per
// 16x16 conv tile per pass. t^2 == t (one-hot) so only 4 planes convolved.
// ---------------------------------------------------------------------------

typedef __attribute__((ext_vector_type(16))) _Float16 v16h;
typedef __attribute__((ext_vector_type(8)))  _Float16 v8h;
typedef __attribute__((ext_vector_type(8)))  float    v8f;

#define BATCH 32
#define NCH   8
#define HW_   512
#define WIN   11
#define OUT_  502          // 512 - 11 + 1
#define TILE  16
#define NTILE 32           // ceil(502/16)
#define HALO  26           // TILE + WIN - 1
#define C1V   1.0e-4f      // (0.01*1.0)^2
#define C2V   9.0e-4f      // (0.03*1.0)^2

#define RAW_PSTRIDE 840    // halves per plane: 26 rows * 32 cols + 8 pad (bank spread)
#define MID_PSTRIDE 512    // halves per plane: 16 cols * 32 rows (K padded 26->32)

// Normalized 1D Gaussian, sigma = 1.5, 11 taps (exp(-d^2/4.5) / sum).
__device__ __constant__ float GW[WIN] = {
  0.00102839f, 0.00759876f, 0.03600077f, 0.10936069f, 0.21300553f,
  0.26601172f, 0.21300553f, 0.10936069f, 0.03600077f, 0.00759876f, 0.00102839f
};

__device__ __forceinline__ float bw(int d) {
  return (d >= 0 && d < WIN) ? GW[d] : 0.0f;
}

__global__ void ssim_zero(float* __restrict__ gacc) {
  gacc[threadIdx.x] = 0.0f;                         // 256 entries: [B, C]
}

__global__ __launch_bounds__(256) void ssim_main(const float* __restrict__ pred,
                                                 const long long* __restrict__ tgt,
                                                 float* __restrict__ gacc) {
  // raw: current plane-group (16 planes) of f16 inputs, row-major 26x32 (+pad)
  // mid: horizontal-conv output, col-major per plane (16 cols x 32 rows) f16
  __shared__ __align__(32) _Float16 raw[16 * RAW_PSTRIDE];   // 26,880 B
  __shared__ __align__(32) _Float16 mid[32 * MID_PSTRIDE];   // 32,768 B

  const int tid  = threadIdx.x;
  const int lane = tid & 31;
  // Wave index is uniform across the wave: pin it to an SGPR so the per-row
  // WMMA loop compiles to scalar branches (no exec-mask juggling around WMMA;
  // WMMA requires EXEC == all ones).
  const int wave = __builtin_amdgcn_readfirstlane(tid >> 5);
  const int x0   = blockIdx.x * TILE;
  const int y0   = blockIdx.y * TILE;
  const int b    = blockIdx.z;

  // Zero mid: pad rows K=26..31 must be 0 for the vertical pass.
  {
    uint4 z = make_uint4(0u, 0u, 0u, 0u);
    uint4* m4 = (uint4*)mid;
    #pragma unroll
    for (int i = 0; i < 8; ++i) m4[tid + i * 256] = z;       // 2048 uint4 total
  }

  // Horizontal banded B fragment: B[K=32][N=16], B[k][j] = w[k-j].
  // 16-bit B layout: lanes 0-15: e<8 -> K=e, e>=8 -> K=e+8; lanes 16-31: +8.
  v16h hB;
  {
    const int j  = lane & 15;
    const int kb = (lane < 16) ? 0 : 8;
    #pragma unroll
    for (int e = 0; e < 16; ++e) hB[e] = (_Float16)bw(((e < 8) ? (kb + e) : (kb + 8 + e)) - j);
  }
  // Vertical banded A fragment: A[M=16][K=32], A[i][k] = w[k-i].
  // A in contiguous K order (same as bm): lanes 0-15 hold row M=lane, K=e; lanes 16-31 row M=lane-16, K=16+e.
  v16h vA;
  {
    const int i  = lane & 15;
    const int kb = (lane < 16) ? 0 : 16;
    #pragma unroll
    for (int e = 0; e < 16; ++e) {
      const int K = kb + e;
      vA[e] = (_Float16)bw(K - i);
    }
  }

  // Two plane-groups through the same raw buffer: g=0 -> {p, p^2}, g=1 -> {t, p*t}.
  for (int g = 0; g < 2; ++g) {
    // -------- phase 1: halo load + softmax + quantity planes (f16 -> LDS) ----
    for (int idx = tid; idx < HALO * 32; idx += 256) {
      const int py = idx >> 5, px = idx & 31;
      const int gy = y0 + py, gx = x0 + px;
      const bool valid = (px < HALO) && (gy < HW_) && (gx < HW_);
      float p[NCH];
      long long tv = -1;
      if (valid) {
        float x[NCH], m = -1e30f;
        #pragma unroll
        for (int c = 0; c < NCH; ++c) {
          x[c] = pred[(((size_t)b * NCH + c) * HW_ + gy) * HW_ + gx];
          m = fmaxf(m, x[c]);
        }
        float s = 0.0f;
        #pragma unroll
        for (int c = 0; c < NCH; ++c) { p[c] = __expf(x[c] - m); s += p[c]; }
        const float inv = 1.0f / s;
        #pragma unroll
        for (int c = 0; c < NCH; ++c) p[c] *= inv;
        tv = tgt[((size_t)b * HW_ + gy) * HW_ + gx];
      } else {
        #pragma unroll
        for (int c = 0; c < NCH; ++c) p[c] = 0.0f;
      }
      const int off = py * 32 + px;
      if (g == 0) {
        #pragma unroll
        for (int c = 0; c < NCH; ++c) {
          raw[c * RAW_PSTRIDE + off]       = (_Float16)p[c];
          raw[(8 + c) * RAW_PSTRIDE + off] = (_Float16)(p[c] * p[c]);
        }
      } else {
        #pragma unroll
        for (int c = 0; c < NCH; ++c) {
          const float tc = (tv == (long long)c) ? 1.0f : 0.0f;
          raw[c * RAW_PSTRIDE + off]       = (_Float16)tc;
          raw[(8 + c) * RAW_PSTRIDE + off] = (_Float16)(tc * p[c]);
        }
      }
    }
    __syncthreads();

    // -------- phase 2: horizontal conv, one WMMA per input row ---------------
    // A[M=plane][K=col]: lanes 0-15 read cols {0..7,16..23}, lanes 16-31 {8..15,24..31}.
    // Loop bound is scalar (wave is SGPR) -> no exec masking around the WMMA.
    for (int r = wave; r < HALO; r += 8) {
      const _Float16* base = raw + (lane & 15) * RAW_PSTRIDE + r * 32 + ((lane < 16) ? 0 : 8);
      const v8h lo = *(const v8h*)base;
      const v8h hi = *(const v8h*)(base + 16);
      const v16h a = __builtin_shufflevector(lo, hi,
                       0, 1, 2, 3, 4, 5, 6, 7, 8, 9, 10, 11, 12, 13, 14, 15);
      v8f c = {};
      c = __builtin_amdgcn_wmma_f32_16x16x32_f16(false, a, false, hB, (short)0, c,
                                                 false, false);
      const int prow = (lane < 16) ? 0 : 8;        // C: M = v (+8 upper half)
      const int j    = lane & 15;                  // C: N = lane&15
      #pragma unroll
      for (int v = 0; v < 8; ++v)
        mid[(g * 16 + prow + v) * MID_PSTRIDE + j * 32 + r] = (_Float16)c[v];
    }
    __syncthreads();
  }

  // -------- phase 3: vertical conv; wave w owns channel w's 4 quantities -----
  // B fragment is one contiguous 32B run per lane (mid is col-major per plane).
  v8f acc[4];
  #pragma unroll
  for (int q = 0; q < 4; ++q) {
    const int pl = q * 8 + wave;                   // q0=p, q1=p^2, q2=t, q3=pt
    const v16h bm = *(const v16h*)(mid + pl * MID_PSTRIDE + (lane & 15) * 32 +
                                   ((lane < 16) ? 0 : 16));
    v8f c = {};
    acc[q] = __builtin_amdgcn_wmma_f32_16x16x32_f16(false, vA, false, bm, (short)0, c,
                                                    false, false);
  }

  // -------- phase 4: SSIM map in registers + masked reduction ----------------
  {
    const int i0 = (lane < 16) ? 0 : 8;
    const int j  = lane & 15;
    float sum = 0.0f;
    #pragma unroll
    for (int v = 0; v < 8; ++v) {
      const float mean1  = acc[0][v];              // conv(p)
      const float e1     = acc[1][v];              // conv(p^2)
      const float mean2  = acc[2][v];              // conv(t)   (== conv(t^2))
      const float ecov   = acc[3][v];              // conv(p*t)
      const float mean12 = mean1 * mean2;
      const float m1sq   = mean1 * mean1;
      const float m2sq   = mean2 * mean2;
      const float var1   = e1 - m1sq;
      const float var2   = mean2 - m2sq;
      const float covar  = ecov - mean12;
      const float cs = (2.0f * covar + C2V) / (var1 + var2 + C2V);
      const float ss = (2.0f * mean12 + C1V) / (m1sq + m2sq + C1V) * cs;
      const bool valid = ((y0 + i0 + v) < OUT_) && ((x0 + j) < OUT_);
      sum += valid ? ss : 0.0f;
    }
    #pragma unroll
    for (int m = 16; m >= 1; m >>= 1) sum += __shfl_xor(sum, m, 32);
    if (lane == 0) atomicAdd(&gacc[b * NCH + wave], sum);   // wave == channel
  }
}

__global__ __launch_bounds__(256) void ssim_final(const float* __restrict__ gacc,
                                                  float* __restrict__ out) {
  __shared__ float red[8];
  const int tid = threadIdx.x, lane = tid & 31, wave = tid >> 5;
  float v = gacc[tid] * (1.0f / (float)(OUT_ * OUT_));      // spatial mean per (b,c)
  v = fmaxf(v, 0.0f);                                       // relu
  #pragma unroll
  for (int m = 16; m >= 1; m >>= 1) v += __shfl_xor(v, m, 32);
  if (lane == 0) red[wave] = v;
  __syncthreads();
  if (tid == 0) {
    float s = 0.0f;
    #pragma unroll
    for (int w = 0; w < 8; ++w) s += red[w];
    out[0] = 1.0f - s * (1.0f / 256.0f);                    // mean over B*C
  }
}

extern "C" void kernel_launch(void* const* d_in, const int* in_sizes, int n_in,
                              void* d_out, int out_size, void* d_ws, size_t ws_size,
                              hipStream_t stream) {
  (void)in_sizes; (void)n_in; (void)out_size; (void)ws_size;
  const float*     pred = (const float*)d_in[0];
  const long long* tgt  = (const long long*)d_in[1];
  float* out  = (float*)d_out;
  float* gacc = (float*)d_ws;                     // 256 floats of scratch

  ssim_zero<<<1, 256, 0, stream>>>(gacc);
  dim3 grid(NTILE, NTILE, BATCH);
  ssim_main<<<grid, 256, 0, stream>>>(pred, tgt, gacc);
  ssim_final<<<1, 256, 0, stream>>>(gacc, out);
}
